// SoftTopKSAE_3994319585727
// MI455X (gfx1250) — matrix operand order repaired and checked by reading, not compiled, not run on hardware
//
#include <hip/hip_runtime.h>
#include <hip/hip_bf16.h>
#include <math.h>

// ---------------------------------------------------------------------------
// SoftTopK SAE forward for MI455X (gfx1250, wave32, WMMA f32_16x16x32_f16,
// async global->LDS staging with ASYNCcnt double buffering)
// B=4096, D_ACT=768, D_DICT=16384, K=64
// ---------------------------------------------------------------------------

#define B_SZ    4096
#define D_ACT   768
#define D_DICT  16384
#define K_TOP   64

typedef __attribute__((ext_vector_type(16))) _Float16 v16h;
typedef __attribute__((ext_vector_type(8)))  _Float16 v8h;
typedef __attribute__((ext_vector_type(8)))  float    v8f;

// ---- async global->LDS (CDNA5) with synchronous fallback ------------------
#if __has_builtin(__builtin_amdgcn_global_load_async_to_lds_b128) && \
    __has_builtin(__builtin_amdgcn_s_wait_asynccnt)
#define HAVE_ASYNC_LDS 1
#define ASYNC_WAIT(n) __builtin_amdgcn_s_wait_asynccnt(n)
#else
#define HAVE_ASYNC_LDS 0
#define ASYNC_WAIT(n) ((void)0)
#endif

// Builtin signature (from hipcc diagnostic): param0 is
//   'int __attribute__((__vector_size__(4*sizeof(int)))) __device__ *'
// i.e. pointer to int4 in AS(1); by symmetry param1 is int4 in AS(3).
typedef int v4i __attribute__((vector_size(16)));
typedef __attribute__((address_space(1))) v4i v4i_glob;
typedef __attribute__((address_space(3))) v4i v4i_lds;

__device__ __forceinline__ void copy16_to_lds(const _Float16* g, _Float16* l) {
#if HAVE_ASYNC_LDS
    __builtin_amdgcn_global_load_async_to_lds_b128(
        (v4i_glob*)g, (v4i_lds*)l, 0, 0);
#else
    *(uint4*)l = *(const uint4*)g;
#endif
}

// ---- WMMA fragment helpers (contiguous k order, same in A and B) -----------------
// A 16x32 f16: lane<16 -> K {0..15}; lane>=16 -> K {16..31}
__device__ __forceinline__ v16h load_a_frag(const _Float16* rowptr, int half) {
    return *(const v16h*)(rowptr + half * 16);
}
// B 32x16 f16: lane holds 16 consecutive K of column N=lane%16 (contiguous 32B)
__device__ __forceinline__ v16h load_b_frag(const _Float16* rowptr, int half) {
    return *(const v16h*)(rowptr + half * 16);
}

__device__ __forceinline__ float reduce16(float v) {
    v += __shfl_xor(v, 1, 32);
    v += __shfl_xor(v, 2, 32);
    v += __shfl_xor(v, 4, 32);
    v += __shfl_xor(v, 8, 32);
    return v;
}

// ---------------------------------------------------------------------------
// Kernel 1: fp32 -> f16 conversions, xc = x - b_dec, zero kdot accumulator
// ---------------------------------------------------------------------------
__global__ __launch_bounds__(256) void sae_prep(
    const float* __restrict__ x, const float* __restrict__ b_dec,
    const float* __restrict__ W_enc, const float* __restrict__ Wk1,
    const float* __restrict__ W_dec,
    _Float16* __restrict__ xh, _Float16* __restrict__ We,
    _Float16* __restrict__ Wk, _Float16* __restrict__ Wd,
    float* __restrict__ kdot)
{
    size_t i = (size_t)blockIdx.x * 256 + threadIdx.x;
    const size_t NX = (size_t)B_SZ * D_ACT;
    const size_t NW = (size_t)D_DICT * D_ACT;
    if (i < NX) xh[i] = (_Float16)(x[i] - b_dec[i % D_ACT]);
    if (i < NW) {
        We[i] = (_Float16)W_enc[i];
        Wk[i] = (_Float16)Wk1[i];
        Wd[i] = (_Float16)W_dec[i];
    }
    if (i < B_SZ) kdot[i] = 0.f;
}

// ---------------------------------------------------------------------------
// Kernel 2: fused encoder.  Block tile 128(M) x 64(N), K=768 in steps of 32,
// double-buffered async LDS staging.
//   acts = relu(xc @ W_enc^T + b_enc)           -> global fp32
//   kdot[row] += sum_col relu(xc @ Wk1^T + bk1)[row,col] * Wk2[col]
// 8 waves, each a 32x32 sub-tile, two accumulator sets (E and H).
// ---------------------------------------------------------------------------
#define ETM 128
#define ETN 64
#define E_NK (D_ACT / 32)   // 24 K-steps
__global__ __launch_bounds__(256) void sae_encode(
    const _Float16* __restrict__ xh, const _Float16* __restrict__ We,
    const _Float16* __restrict__ Wk, const float* __restrict__ b_enc,
    const float* __restrict__ bk1, const float* __restrict__ Wk2,
    float* __restrict__ acts, float* __restrict__ kdot)
{
    __shared__ _Float16 At[2][ETM * 32];
    __shared__ _Float16 Be[2][ETN * 32];
    __shared__ _Float16 Bk[2][ETN * 32];
    __shared__ float    rowsum[ETM];

    const int tid  = threadIdx.x;
    const int wave = tid >> 5, lane = tid & 31;
    const int half = lane >> 4, l16 = lane & 15;
    const int m0 = blockIdx.y * ETM;
    const int n0 = blockIdx.x * ETN;
    const int wm = (wave >> 1) * 32;   // 0,32,64,96
    const int wn = (wave & 1) * 32;    // 0,32

    v8f accE[2][2] = {};
    v8f accH[2][2] = {};

    for (int i = tid; i < ETM; i += 256) rowsum[i] = 0.f;

    // 16B-chunk staging: A = 512 chunks (2/thread), Be/Bk = 256 chunks (1/thread)
    auto stage = [&](int buf, int kk) {
#pragma unroll
        for (int i = 0; i < 2; ++i) {
            const int c = tid + i * 256;            // chunk 0..511
            const int r = c >> 2, cc = (c & 3) * 8; // 4 chunks per 32-half row
            copy16_to_lds(&xh[(size_t)(m0 + r) * D_ACT + kk + cc], &At[buf][c * 8]);
        }
        {
            const int r = tid >> 2, cc = (tid & 3) * 8;
            copy16_to_lds(&We[(size_t)(n0 + r) * D_ACT + kk + cc], &Be[buf][tid * 8]);
            copy16_to_lds(&Wk[(size_t)(n0 + r) * D_ACT + kk + cc], &Bk[buf][tid * 8]);
        }
    };

    stage(0, 0);
    for (int ks = 0; ks < E_NK; ++ks) {
        const int buf = ks & 1;
        if (ks + 1 < E_NK) {
            stage(buf ^ 1, (ks + 1) * 32);  // overlap DMA of next tile w/ compute
            ASYNC_WAIT(4);                  // the 4 older ops (stage ks) landed
        } else {
            ASYNC_WAIT(0);
        }
        __syncthreads();                    // all waves' stage-ks data visible

        v16h af[2], be_f[2], bk_f[2];
#pragma unroll
        for (int mi = 0; mi < 2; ++mi)
            af[mi] = load_a_frag(&At[buf][(wm + mi * 16 + l16) * 32], half);
#pragma unroll
        for (int ni = 0; ni < 2; ++ni) {
            be_f[ni] = load_b_frag(&Be[buf][(wn + ni * 16 + l16) * 32], half);
            bk_f[ni] = load_b_frag(&Bk[buf][(wn + ni * 16 + l16) * 32], half);
        }
#pragma unroll
        for (int mi = 0; mi < 2; ++mi)
#pragma unroll
            for (int ni = 0; ni < 2; ++ni) {
                accE[mi][ni] = __builtin_amdgcn_wmma_f32_16x16x32_f16(
                    false, af[mi], false, be_f[ni], (short)0, accE[mi][ni], false, false);
                accH[mi][ni] = __builtin_amdgcn_wmma_f32_16x16x32_f16(
                    false, af[mi], false, bk_f[ni], (short)0, accH[mi][ni], false, false);
            }
        __syncthreads();                    // buffer safe to overwrite at ks+2
    }

    // ---- acts epilogue: relu(+b_enc), store fp32 ----
#pragma unroll
    for (int mi = 0; mi < 2; ++mi)
#pragma unroll
        for (int ni = 0; ni < 2; ++ni) {
            const int col = n0 + wn + ni * 16 + l16;
            const float be = b_enc[col];
#pragma unroll
            for (int r = 0; r < 8; ++r) {
                const int row = m0 + wm + mi * 16 + r + half * 8;
                float a = accE[mi][ni][r] + be;
                acts[(size_t)row * D_DICT + col] = a > 0.f ? a : 0.f;
            }
        }

    // ---- k-estimator partial dot: relu(H)·Wk2, reduce across 16 lanes ----
#pragma unroll
    for (int mi = 0; mi < 2; ++mi)
#pragma unroll
        for (int r = 0; r < 8; ++r) {
            float contrib = 0.f;
#pragma unroll
            for (int ni = 0; ni < 2; ++ni) {
                const int col = n0 + wn + ni * 16 + l16;
                float h = accH[mi][ni][r] + bk1[col];
                h = h > 0.f ? h : 0.f;
                contrib += h * Wk2[col];
            }
            contrib = reduce16(contrib);
            if (l16 == 0)
                atomicAdd(&rowsum[wm + mi * 16 + r + half * 8], contrib); // ds_add_f32
        }
    __syncthreads();
    if (tid < ETM) atomicAdd(&kdot[m0 + tid], rowsum[tid]);
}

// ---------------------------------------------------------------------------
// Kernel 3: per-row k_est + exact top-m threshold via 4-pass radix select
// (acts >= 0 so float bits order like uints), then write masked f16 enc.
// One 256-thread block per row.
// ---------------------------------------------------------------------------
__global__ __launch_bounds__(256) void sae_select(
    const float* __restrict__ acts, const float* __restrict__ kdot,
    const float* __restrict__ bk2, const int* __restrict__ kptr,
    _Float16* __restrict__ enc)
{
    __shared__ unsigned hist[256];
    __shared__ unsigned s_prefix, s_m;

    const int b   = blockIdx.x;
    const int tid = threadIdx.x;
    const float* row = acts + (size_t)b * D_DICT;

    const float z  = kdot[b] + bk2[0];
    const float ke = (2.0f * (float)kptr[0]) / (1.0f + expf(-z));
    int m = (int)ceilf(ke);
    if (m < 1) m = 1;
    if (m > D_DICT) m = D_DICT;

    unsigned prefix = 0u, mrem = (unsigned)m;
#pragma unroll
    for (int pass = 0; pass < 4; ++pass) {
        const int shift = 24 - pass * 8;
        const unsigned hmask = (pass == 0) ? 0u : (0xFFFFFFFFu << (shift + 8));
        hist[tid] = 0u;
        __syncthreads();
        for (int i = tid; i < D_DICT; i += 256) {
            unsigned key = __float_as_uint(row[i]);
            if ((key & hmask) == prefix)
                atomicAdd(&hist[(key >> shift) & 255u], 1u);
        }
        __syncthreads();
        if (tid == 0) {
            unsigned cum = 0; int bin = 255;
            for (; bin >= 0; --bin) {
                unsigned c = hist[bin];
                if (cum + c >= mrem) break;
                cum += c;
            }
            if (bin < 0) bin = 0;
            s_prefix = prefix | ((unsigned)bin << shift);
            s_m = mrem - cum;
        }
        __syncthreads();
        prefix = s_prefix;
        mrem   = s_m;
        __syncthreads();
    }

    const float thresh = __uint_as_float(prefix);
    for (int i = tid; i < D_DICT; i += 256) {
        float v = row[i];
        float e = (v >= thresh && v > 0.f) ? v : 0.f;
        enc[(size_t)b * D_DICT + i] = (_Float16)e;
    }
}

// ---------------------------------------------------------------------------
// Kernel 4: decoder GEMM  x_hat = enc @ W_dec^T + b_dec
// Block tile 64(M) x 128(N), K=16384 in steps of 32, double-buffered async
// staging. 8 waves -> 32x32 each.
// ---------------------------------------------------------------------------
#define DTM 64
#define DTN 128
#define D_NK (D_DICT / 32)   // 512 K-steps
__global__ __launch_bounds__(256) void sae_decode(
    const _Float16* __restrict__ enc, const _Float16* __restrict__ Wd,
    const float* __restrict__ b_dec, float* __restrict__ out)
{
    __shared__ _Float16 As[2][DTM * 32];
    __shared__ _Float16 Bs[2][DTN * 32];

    const int tid  = threadIdx.x;
    const int wave = tid >> 5, lane = tid & 31;
    const int half = lane >> 4, l16 = lane & 15;
    const int m0 = blockIdx.y * DTM;
    const int n0 = blockIdx.x * DTN;
    const int wm = (wave >> 2) * 32;   // 0,32
    const int wn = (wave & 3) * 32;    // 0..96

    v8f acc[2][2] = {};

    // A = 256 chunks (1/thread), B = 512 chunks (2/thread)
    auto stage = [&](int buf, int kk) {
        {
            const int r = tid >> 2, cc = (tid & 3) * 8;
            copy16_to_lds(&enc[(size_t)(m0 + r) * D_DICT + kk + cc], &As[buf][tid * 8]);
        }
#pragma unroll
        for (int i = 0; i < 2; ++i) {
            const int c = tid + i * 256;
            const int r = c >> 2, cc = (c & 3) * 8;
            copy16_to_lds(&Wd[(size_t)(n0 + r) * D_DICT + kk + cc], &Bs[buf][c * 8]);
        }
    };

    stage(0, 0);
    for (int ks = 0; ks < D_NK; ++ks) {
        const int buf = ks & 1;
        if (ks + 1 < D_NK) {
            stage(buf ^ 1, (ks + 1) * 32);
            ASYNC_WAIT(3);
        } else {
            ASYNC_WAIT(0);
        }
        __syncthreads();

        v16h af[2], bf[2];
#pragma unroll
        for (int mi = 0; mi < 2; ++mi)
            af[mi] = load_a_frag(&As[buf][(wm + mi * 16 + l16) * 32], half);
#pragma unroll
        for (int ni = 0; ni < 2; ++ni)
            bf[ni] = load_b_frag(&Bs[buf][(wn + ni * 16 + l16) * 32], half);
#pragma unroll
        for (int mi = 0; mi < 2; ++mi)
#pragma unroll
            for (int ni = 0; ni < 2; ++ni)
                acc[mi][ni] = __builtin_amdgcn_wmma_f32_16x16x32_f16(
                    false, af[mi], false, bf[ni], (short)0, acc[mi][ni], false, false);
        __syncthreads();
    }

#pragma unroll
    for (int mi = 0; mi < 2; ++mi)
#pragma unroll
        for (int ni = 0; ni < 2; ++ni) {
            const int col = n0 + wn + ni * 16 + l16;
            const float bd = b_dec[col];
#pragma unroll
            for (int r = 0; r < 8; ++r) {
                const int row = m0 + wm + mi * 16 + r + half * 8;
                out[(size_t)row * D_ACT + col] = acc[mi][ni][r] + bd;
            }
        }
}

// ---------------------------------------------------------------------------
// Launch
// ---------------------------------------------------------------------------
extern "C" void kernel_launch(void* const* d_in, const int* in_sizes, int n_in,
                              void* d_out, int out_size, void* d_ws, size_t ws_size,
                              hipStream_t stream)
{
    (void)in_sizes; (void)n_in; (void)out_size; (void)ws_size;

    const float* x     = (const float*)d_in[0];
    const float* W_enc = (const float*)d_in[1];
    const float* b_enc = (const float*)d_in[2];
    const float* W_dec = (const float*)d_in[3];
    const float* b_dec = (const float*)d_in[4];
    const float* Wk1   = (const float*)d_in[5];
    const float* bk1   = (const float*)d_in[6];
    const float* Wk2   = (const float*)d_in[7];
    const float* bk2   = (const float*)d_in[8];
    const int*   kptr  = (const int*)d_in[9];
    float* out = (float*)d_out;

    // workspace layout (bytes)
    constexpr size_t SZ_XH   = (size_t)B_SZ * D_ACT * 2;     //   6 MB
    constexpr size_t SZ_W16  = (size_t)D_DICT * D_ACT * 2;   //  24 MB each
    constexpr size_t SZ_ACTS = (size_t)B_SZ * D_DICT * 4;    // 256 MB
    constexpr size_t SZ_ENC  = (size_t)B_SZ * D_DICT * 2;    // 128 MB

    char* w = (char*)d_ws;
    size_t off = 0;
    _Float16* xh   = (_Float16*)(w + off); off += SZ_XH;
    _Float16* We   = (_Float16*)(w + off); off += SZ_W16;
    _Float16* Wk   = (_Float16*)(w + off); off += SZ_W16;
    _Float16* Wd   = (_Float16*)(w + off); off += SZ_W16;
    float*    acts = (float*)   (w + off); off += SZ_ACTS;
    _Float16* enc  = (_Float16*)(w + off); off += SZ_ENC;
    float*    kdot = (float*)   (w + off);

    // 1) convert + center + zero accumulators
    {
        const size_t nmax = (size_t)D_DICT * D_ACT;
        dim3 grid((unsigned)((nmax + 255) / 256));
        sae_prep<<<grid, 256, 0, stream>>>(x, b_dec, W_enc, Wk1, W_dec,
                                           xh, We, Wk, Wd, kdot);
    }
    // 2) fused encoder GEMM + k-estimator dot
    {
        dim3 grid(D_DICT / ETN, B_SZ / ETM);   // (256, 32)
        sae_encode<<<grid, 256, 0, stream>>>(xh, We, Wk, b_enc, bk1, Wk2,
                                             acts, kdot);
    }
    // 3) per-row threshold select + masked f16 enc
    {
        sae_select<<<dim3(B_SZ), 256, 0, stream>>>(acts, kdot, bk2, kptr, enc);
    }
    // 4) decoder GEMM + b_dec
    {
        dim3 grid(D_ACT / DTN, B_SZ / DTM);    // (6, 64)
        sae_decode<<<grid, 256, 0, stream>>>(enc, Wd, b_dec, out);
    }
}
